// SeqPredictor_31937376813586
// MI455X (gfx1250) — compile-verified, matrix operand order checked by reading, not run on hardware
//
#include <hip/hip_runtime.h>

typedef float  v2f   __attribute__((ext_vector_type(2)));
typedef float  v8f   __attribute__((ext_vector_type(8)));
typedef __bf16 v16bf __attribute__((ext_vector_type(16)));

#define BATCH    16
#define N_RIGIDS 8192
#define N_RES    2048
#define C_FRAME  384
#define C_S      384
#define N_AA     21
#define LN_EPS   1e-5f

// f32 LDS row stride (kernel2): 388 % 64 == 4 -> rows land on distinct banks
#define LDSSF 388
// bf16 LDS row stride (kernel1): 392 halves = 784 B (16B-aligned rows; 196 words
// per row -> 196 % 64 == 4 -> b128 reads from 16 lanes cover all 64 banks)
#define LDSSB 392

#define SEQ_ELEMS (BATCH * N_RES * C_S)   // fp32 accumulator elements in ws

__device__ __forceinline__ unsigned short f2bf(float f) {
  unsigned int u = __float_as_uint(f);
  u += 0x7FFFu + ((u >> 16) & 1u);       // round-to-nearest-even
  return (unsigned short)(u >> 16);
}

union FragBF {                            // 16 bf16 = two 16B halves
  v16bf v;
  uint4 q[2];
};

// ---------------------------------------------------------------------------
// Kernel 0a: seed workspace accumulator with `out` (float4 copy)
// ---------------------------------------------------------------------------
__global__ void init_ws_kernel(const float4* __restrict__ src,
                               float4* __restrict__ dst, int n4) {
  int i = blockIdx.x * blockDim.x + threadIdx.x;
  if (i < n4) dst[i] = src[i];
}

// ---------------------------------------------------------------------------
// Kernel 0b: W_su f32 [k][n] -> bf16 transposed [n][k] (B-fragment friendly)
// ---------------------------------------------------------------------------
__global__ void convert_wsu_kernel(const float* __restrict__ W_su,
                                   unsigned short* __restrict__ Wb) {
  int i = blockIdx.x * blockDim.x + threadIdx.x;   // i = n*C_FRAME + k
  if (i < C_FRAME * C_S) {
    int n = i / C_FRAME, k = i - n * C_FRAME;
    Wb[i] = f2bf(W_su[k * C_S + n]);
  }
}

// ---------------------------------------------------------------------------
// Kernel 1: fused LayerNorm + bf16 GEMM1 + masked atomic scatter-add.
// One block = 32 rigid rows of one batch. 256 threads = 8 waves.
// Wave w: N-tiles 3w..3w+2 (48 cols) x both 16-row M-tiles (6 accumulators).
// ---------------------------------------------------------------------------
__global__ __launch_bounds__(256) void ln_gemm_scatter_kernel(
    const float* __restrict__ x,     // [B, N_RIGIDS, C_FRAME]
    const int*   __restrict__ idx,   // [B, N_RIGIDS]
    const float* __restrict__ mask,  // [B, N_RIGIDS]
    const float* __restrict__ ln_g,
    const float* __restrict__ ln_b,
    const unsigned short* __restrict__ Wb,  // bf16 [n][k]
    const float* __restrict__ b_su,
    float* __restrict__ ws)          // [B, N_RES, C_S] fp32 accumulator
{
  __shared__ unsigned short s_a[32 * LDSSB];  // bf16 LN output, row-major
  __shared__ int   s_idx[32];
  __shared__ float s_mask[32];

  const int tid  = threadIdx.x;
  const int wave = tid >> 5;
  const int lane = tid & 31;
  const int half = lane >> 4;
  const int sub  = lane & 15;

  const int  blocksPerBatch = N_RIGIDS / 32;
  const int  batch  = blockIdx.x / blocksPerBatch;
  const int  mblock = blockIdx.x % blocksPerBatch;
  const long rigid0 = (long)batch * N_RIGIDS + (long)mblock * 32;

  if (tid < 32) {
    s_idx[tid]  = idx[rigid0 + tid];
    s_mask[tid] = mask[rigid0 + tid];
  }

  // ---- LayerNorm (fp32) -> bf16 into LDS. Wave w: rows 4w..4w+3 ----
#pragma unroll
  for (int pass = 0; pass < 2; ++pass) {
    const int    row = wave * 4 + pass * 2 + half;
    const float* xr  = x + (rigid0 + row) * C_FRAME;
    float v[24];
    float s = 0.f, sq = 0.f;
#pragma unroll
    for (int i = 0; i < 24; ++i) {
      float t = xr[sub + 16 * i];
      v[i] = t; s += t; sq += t * t;
    }
#pragma unroll
    for (int m = 1; m < 16; m <<= 1) {     // xor<16 stays within 16-lane group
      s  += __shfl_xor(s,  m, 32);
      sq += __shfl_xor(sq, m, 32);
    }
    const float mu  = s  * (1.0f / C_FRAME);
    const float var = sq * (1.0f / C_FRAME) - mu * mu;
    const float rs  = rsqrtf(var + LN_EPS);
#pragma unroll
    for (int i = 0; i < 24; ++i) {
      const int f = sub + 16 * i;
      s_a[row * LDSSB + f] = f2bf((v[i] - mu) * rs * ln_g[f] + ln_b[f]);
    }
  }
  __syncthreads();

  // ---- bf16 WMMA GEMM: D = A(16x32) x B(32x16), K-loop of 12 ----
  v8f acc[6] = {};                       // [ntile][mtile]: acc[2t+mt]
  const int nbase = wave * 48;
  const int kb0   = half * 8;            // ISA 16-bit layout: VGPR0-3 = K 8h+0..7
  const unsigned short* sA0 = s_a + (unsigned)sub        * LDSSB;
  const unsigned short* sA1 = s_a + (unsigned)(16 + sub) * LDSSB;

  for (int kb = 0; kb < C_FRAME; kb += 32) {
    FragBF a0, a1;
    a0.q[0] = *(const uint4*)(sA0 + kb + kb0);        // ds_load_b128
    a0.q[1] = *(const uint4*)(sA0 + kb + kb0 + 16);
    a1.q[0] = *(const uint4*)(sA1 + kb + kb0);
    a1.q[1] = *(const uint4*)(sA1 + kb + kb0 + 16);
#pragma unroll
    for (int t = 0; t < 3; ++t) {
      const unsigned short* wr =
          Wb + (long)(nbase + t * 16 + sub) * C_FRAME + kb + kb0;
      FragBF b;
      b.q[0] = *(const uint4*)(wr);                   // global_load_b128
      b.q[1] = *(const uint4*)(wr + 16);
      acc[2 * t + 0] = __builtin_amdgcn_wmma_f32_16x16x32_bf16(
          false, a0.v, false, b.v, (short)0, acc[2 * t + 0], false, false);
      acc[2 * t + 1] = __builtin_amdgcn_wmma_f32_16x16x32_bf16(
          false, a1.v, false, b.v, (short)0, acc[2 * t + 1], false, false);
    }
  }

  // ---- Epilogue: bias + mask, atomic scatter-add into per-residue slots ----
  float* wsB = ws + (long)batch * N_RES * C_S;
#pragma unroll
  for (int t = 0; t < 3; ++t) {
    const int   col = nbase + t * 16 + sub;
    const float bs  = b_su[col];
#pragma unroll
    for (int mt = 0; mt < 2; ++mt) {
      const v8f a = acc[2 * t + mt];
#pragma unroll
      for (int r = 0; r < 8; ++r) {
        const int m = mt * 16 + r + 8 * half;   // C/D layout: VGPR r -> M=r(+8)
        unsafeAtomicAdd(wsB + (long)s_idx[m] * C_S + col,
                        (a[r] + bs) * s_mask[m]);
      }
    }
  }
}

// ---------------------------------------------------------------------------
// Kernel 2: logits = seq_embed @ W_out + b_out  (fp32 WMMA 16x16x4)
// One block = 128 rows (8 waves x 16-row M-tiles). N=21 -> 2 N-tiles.
// ---------------------------------------------------------------------------
__global__ __launch_bounds__(256) void out_gemm_kernel(
    const float* __restrict__ ws,     // [B*N_RES, C_S]
    const float* __restrict__ W_out,  // [C_S, N_AA]
    const float* __restrict__ b_out,
    float* __restrict__ out)          // [B*N_RES, N_AA]
{
  __shared__ float s_wt[32 * LDSSF];  // W_out^T, zero-padded cols 21..31

  const int tid  = threadIdx.x;
  const int wave = tid >> 5;
  const int lane = tid & 31;
  const int half = lane >> 4;
  const int sub  = lane & 15;

  for (int i = tid; i < C_S * 32; i += 256) {
    const int n = i & 31;
    const int k = i >> 5;
    s_wt[n * LDSSF + k] = (n < N_AA) ? W_out[k * N_AA + n] : 0.0f;
  }
  __syncthreads();

  const long   row0 = (long)blockIdx.x * 128 + wave * 16;
  const float* Ar   = ws + (row0 + sub) * C_S;
  const float* wt0  = &s_wt[sub * LDSSF];
  const float* wt1  = &s_wt[(16 + sub) * LDSSF];

  v8f acc0 = {}, acc1 = {};
  for (int kb = 0; kb < C_S; kb += 4) {
    const int k = kb + 2 * half;            // f32 A layout: v0=K{0,2}, v1=K{1,3}
    v2f a  = *(const v2f*)&Ar[k];
    v2f b0 = *(const v2f*)&wt0[k];
    v2f b1 = *(const v2f*)&wt1[k];
    acc0 = __builtin_amdgcn_wmma_f32_16x16x4_f32(false, a, false, b0, (short)0, acc0, false, false);
    acc1 = __builtin_amdgcn_wmma_f32_16x16x4_f32(false, a, false, b1, (short)0, acc1, false, false);
  }

  const float bo0 = b_out[sub];
  const int   c1  = 16 + sub;
  const float bo1 = (c1 < N_AA) ? b_out[c1] : 0.0f;
#pragma unroll
  for (int r = 0; r < 8; ++r) {
    const int m = r + 8 * half;
    float* o = out + (row0 + m) * N_AA;
    o[sub] = acc0[r] + bo0;
    if (c1 < N_AA) o[c1] = acc1[r] + bo1;
  }
}

// ---------------------------------------------------------------------------
extern "C" void kernel_launch(void* const* d_in, const int* in_sizes, int n_in,
                              void* d_out, int out_size, void* d_ws, size_t ws_size,
                              hipStream_t stream) {
  const float* x     = (const float*)d_in[0];  // rigids_embed_flat
  const int*   idx   = (const int*)  d_in[1];  // rigids_to_res_idx
  const float* mask  = (const float*)d_in[2];  // rigids_mask
  const float* out0  = (const float*)d_in[3];  // out (seed)
  const float* ln_g  = (const float*)d_in[4];
  const float* ln_b  = (const float*)d_in[5];
  const float* W_su  = (const float*)d_in[6];
  const float* b_su  = (const float*)d_in[7];
  const float* W_out = (const float*)d_in[8];
  const float* b_out = (const float*)d_in[9];

  float*          out = (float*)d_out;
  float*          ws  = (float*)d_ws;                      // fp32 accumulator
  unsigned short* Wb  = (unsigned short*)(ws + SEQ_ELEMS); // bf16 W_su^T

  const int n4 = SEQ_ELEMS / 4;
  init_ws_kernel<<<(n4 + 255) / 256, 256, 0, stream>>>(
      (const float4*)out0, (float4*)ws, n4);

  convert_wsu_kernel<<<(C_FRAME * C_S + 255) / 256, 256, 0, stream>>>(W_su, Wb);

  ln_gemm_scatter_kernel<<<BATCH * (N_RIGIDS / 32), 256, 0, stream>>>(
      x, idx, mask, ln_g, ln_b, Wb, b_su, ws);

  out_gemm_kernel<<<(BATCH * N_RES) / 128, 256, 0, stream>>>(
      ws, W_out, b_out, out);
}
